// Model_34548716929080
// MI455X (gfx1250) — matrix operand order repaired and checked by reading, not compiled, not run on hardware
//
#include <hip/hip_runtime.h>
#include <hip/hip_bf16.h>
#include <math.h>

typedef __attribute__((ext_vector_type(16))) _Float16 v16h;
typedef __attribute__((ext_vector_type(8)))  float    v8f;

#define USE_ASYNC_LDS 1   // gfx1250 GLOBAL_LOAD_ASYNC_TO_LDS for A-tile staging

// ---------------------------------------------------------------------------
// Model constants (match reference)
// ---------------------------------------------------------------------------
#define NIMG   32
#define KCODES 64
#define DDIM   64
#define VQ_ROWS (NIMG * 64 * 64)           // 131072
#define XRE_OFF  1
#define PERP_OFF (1 + NIMG * 256 * 256)    // 2097153
#define QN_OFF   (PERP_OFF + 1)            // 2097154

// All WMMA-GEMM layers in this net share a 64x64 output grid:
#define OGRID   4096                       // 64*64 pixels per image
#define MTOTAL  (NIMG * OGRID)             // 131072 GEMM rows

// ---------------------------------------------------------------------------
// GEMM-conv parameters
// ---------------------------------------------------------------------------
struct GemmP {
  const _Float16* A;      // NHWC f16 activations
  const _Float16* Bp;     // pre-packed B fragments
  const float*    bias;   // f32 bias or nullptr
  const _Float16* Res;    // residual (relu applied in epilogue) or nullptr
  _Float16*       Of16;   // NHWC f16 out (or nullptr)
  float*          Of32;   // NHWC f32 out (or nullptr)
  int IH, IW, Cin, Cout;
  int cinsh;              // log2(Cin)
  int KW;                 // kernel width (mode 0); 1, 3 or 4
  int stride, pad;
  int Ktot;               // padded K, multiple of 32
  int mode;               // 0 = conv; 1 = stride-2 deconv, one parity class
  int py, px;             // parity (mode 1); physical out grid = 128x128
  int relu_a, relu_out;
};

// ---------------------------------------------------------------------------
// WMMA implicit-GEMM convolution.
// Block = 256 threads (8 waves). M-tile = 64 output pixels.
// wave w: M-subtile = w&3, N-group = w>>2; nt = (w>>2) + 2*i, i < NT/2.
// A tile staged memory->LDS with GLOBAL_LOAD_ASYNC_TO_LDS_B128 (ASYNCcnt).
// ---------------------------------------------------------------------------
template<int NT>
__global__ __launch_bounds__(256) void gemm_conv_wmma(GemmP p) {
  __shared__ _Float16 Alds[64 * 32];     // 64 rows x 32 K halves = 4 KB

  const int tid  = threadIdx.x;
  const int lane = tid & 31;
  const int wave = tid >> 5;
  const int msub = wave & 3;
  const int ntb  = wave >> 2;
  const int m0   = blockIdx.x * 64;
  const int KC   = p.Ktot >> 5;

  v8f acc[NT / 2];
#pragma unroll
  for (int i = 0; i < NT / 2; ++i)
#pragma unroll
    for (int j = 0; j < 8; ++j) acc[i][j] = 0.0f;

  // cooperative A-load role: 64 rows x 4 segments of 8 halves
  const int arow = tid >> 2;
  const int aseg = tid & 3;
  const int apix = m0 + arow;
  const int an   = apix >> 12;           // / OGRID
  const int arem = apix & 4095;
  const int aoy  = arem >> 6;            // / 64
  const int aox  = arem & 63;

  for (int kc = 0; kc < KC; ++kc) {
    // ---- stage A chunk (64 rows x 32 K) into LDS ----
    {
      const int k   = (kc << 5) + (aseg << 3);
      const int tap = k >> p.cinsh;
      const int c   = k & (p.Cin - 1);
      int iy, ix;
      if (p.mode == 0) {
        int ky, kx;
        if (p.KW == 4)      { ky = tap >> 2;         kx = tap & 3; }
        else if (p.KW == 3) { ky = (tap * 43) >> 7;  kx = tap - ky * 3; }
        else                { ky = 0;                kx = 0; }
        iy = aoy * p.stride - p.pad + ky;
        ix = aox * p.stride - p.pad + kx;
      } else {                           // stride-2 deconv, parity (py,px)
        iy = aoy + (tap >> 1) - 1 + p.py;
        ix = aox + (tap & 1)  - 1 + p.px;
      }
      const int loff = (arow << 5) + (aseg << 3);
      if (iy >= 0 && iy < p.IH && ix >= 0 && ix < p.IW) {
        const _Float16* gptr =
            p.A + (((size_t)an * p.IH + iy) * p.IW + ix) * p.Cin + c;
#if USE_ASYNC_LDS
        const unsigned ldsoff = (unsigned)(uintptr_t)(&Alds[loff]);
        asm volatile("global_load_async_to_lds_b128 %0, %1, off"
                     :: "v"(ldsoff), "v"(gptr) : "memory");
#else
        *reinterpret_cast<uint4*>(&Alds[loff]) =
            *reinterpret_cast<const uint4*>(gptr);
#endif
      } else {
        *reinterpret_cast<uint4*>(&Alds[loff]) = make_uint4(0u, 0u, 0u, 0u);
      }
    }
#if USE_ASYNC_LDS
    asm volatile("s_wait_asynccnt 0x0" ::: "memory");
#endif
    __syncthreads();

    // ---- build 16x32 A fragment (ISA 7.12.2 16-bit A layout) ----
    union { unsigned u[8]; v16h h; } Af;
    {
      const int r  = (lane & 15) + (msub << 4);
      const int kb = (lane >> 4) << 2;     // dword offset: lanes>=16 -> K+8
      const unsigned* ap = reinterpret_cast<const unsigned*>(&Alds[r << 5]);
#pragma unroll
      for (int vI = 0; vI < 4; ++vI) Af.u[vI]     = ap[kb + vI];       // K 0..7
#pragma unroll
      for (int vI = 0; vI < 4; ++vI) Af.u[4 + vI] = ap[8 + kb + vI];   // K 16..23
      if (p.relu_a) {
#pragma unroll
        for (int q = 0; q < 16; ++q)
          Af.h[q] = (Af.h[q] > (_Float16)0.f) ? Af.h[q] : (_Float16)0.f;
      }
    }

    // ---- WMMA over owned N-tiles ----
#pragma unroll
    for (int i = 0; i < NT / 2; ++i) {
      const int nt = ntb + (i << 1);
      union { uint4 q[2]; v16h h; } Bf;
      const uint4* bp = reinterpret_cast<const uint4*>(
          p.Bp + (((size_t)(kc * NT + nt) << 5) + lane) * 16);
      Bf.q[0] = bp[0];
      Bf.q[1] = bp[1];
      acc[i] = __builtin_amdgcn_wmma_f32_16x16x32_f16(
          false, Af.h, false, Bf.h, (short)0, acc[i], false, false);
    }
    __syncthreads();
  }

  // ---- epilogue: bias + residual + relu + store ----
  const int col0  = lane & 15;
  const int rhalf = (lane >> 4) << 3;
#pragma unroll
  for (int i = 0; i < NT / 2; ++i) {
    const int nt  = ntb + (i << 1);
    const int col = (nt << 4) + col0;
    const float bv = p.bias ? p.bias[col] : 0.0f;
#pragma unroll
    for (int j = 0; j < 8; ++j) {
      const int m = m0 + (msub << 4) + rhalf + j;
      float v = acc[i][j] + bv;
      if (p.Res) {
        float rr = (float)p.Res[(size_t)m * p.Cout + col];
        v += (rr > 0.f) ? rr : 0.f;
      }
      if (p.relu_out) v = fmaxf(v, 0.f);
      size_t oidx;
      if (p.mode == 0) {
        oidx = (size_t)m * p.Cout + col;
      } else {
        const int n  = m >> 12;
        const int rm = m & 4095;
        const int oy = rm >> 6;
        const int ox = rm & 63;
        oidx = (((size_t)n * 128 + (2 * oy + p.py)) * 128
                + (2 * ox + p.px)) * p.Cout + col;
      }
      if (p.Of16) p.Of16[oidx] = (_Float16)v;
      if (p.Of32) p.Of32[oidx] = v;
    }
  }
}

// ---------------------------------------------------------------------------
// Pack f32 weights into f16 B fragments: out[((kc*NT+nt)*32+lane)*16+h].
// B 32x16 layout: lanes 0-15 col N, K 0..7 and 16..23; lanes 16-31 K 8..15 and 24..31.
// mode 0: conv, w is (Cout,Cin,KH,KW)
// mode 1: deconv stride 1, w is (Cin,Cout,KH,KW), kernel flipped
// mode 2: deconv 4x4 stride 2, parity class (py,px), taps = 2x2 subset
// ---------------------------------------------------------------------------
__global__ void pack_b(const float* w, _Float16* out, int Ktot, int NT,
                       int Cin, int Cout, int KH, int KW, int mode,
                       int py, int px) {
  const int idx   = blockIdx.x * blockDim.x + threadIdx.x;
  const int total = Ktot * (NT << 4);
  if (idx >= total) return;
  const int h    = idx & 15;
  const int lane = (idx >> 4) & 31;
  const int rest = idx >> 9;
  const int nt   = rest % NT;
  const int kc   = rest / NT;
  const int n    = (nt << 4) + (lane & 15);
  const int k    = (kc << 5) + ((lane >> 4) << 3) + (h & 7) + ((h >> 3) << 4);
  const int tap  = k / Cin;
  const int c    = k - tap * Cin;
  float val = 0.f;
  if (mode == 0) {
    if (tap < KH * KW) {
      const int ky = tap / KW, kx = tap - (tap / KW) * KW;
      val = w[(((size_t)n * Cin + c) * KH + ky) * KW + kx];
    }
  } else if (mode == 1) {
    if (tap < KH * KW) {
      const int ky = tap / KW, kx = tap - (tap / KW) * KW;
      val = w[(((size_t)c * Cout + n) * KH + (KH - 1 - ky)) * KW + (KW - 1 - kx)];
    }
  } else {                                   // mode 2: 4x4 s2 parity subset
    const int tk = tap >> 1, tx = tap & 1;
    const int ky = py ? (1 + 2 * tk) : (2 * tk);
    const int kx = px ? (1 + 2 * tx) : (2 * tx);
    val = w[(((size_t)c * Cout + n) * 4 + (3 - ky)) * 4 + (3 - kx)];
  }
  out[((size_t)(kc * NT + nt) * 32 + lane) * 16 + h] = (_Float16)val;
}

// ---------------------------------------------------------------------------
// enc_c1: 1->64ch, 4x4, s2, p1, relu.  Thread = (pixel, 4 channels).
// Out NHWC f16 (32,128,128,64).
// ---------------------------------------------------------------------------
__global__ void enc_c1_kernel(const float* __restrict__ x,
                              const float* __restrict__ w,
                              const float* __restrict__ b,
                              _Float16* __restrict__ out) {
  const int idx = blockIdx.x * blockDim.x + threadIdx.x;   // 8,388,608
  const int cg  = (idx & 15) << 2;
  const int pix = idx >> 4;
  const int n   = pix >> 14;
  const int rm  = pix & 16383;
  const int oy  = rm >> 7;
  const int ox  = rm & 127;
  float a0 = b[cg], a1 = b[cg + 1], a2 = b[cg + 2], a3 = b[cg + 3];
#pragma unroll
  for (int ky = 0; ky < 4; ++ky) {
    const int iy = 2 * oy - 1 + ky;
    if (iy < 0 || iy >= 256) continue;
#pragma unroll
    for (int kx = 0; kx < 4; ++kx) {
      const int ix = 2 * ox - 1 + kx;
      if (ix < 0 || ix >= 256) continue;
      const float xv = x[((size_t)n << 16) + (iy << 8) + ix];
      const int wo = ky * 4 + kx;
      a0 += xv * w[(cg + 0) * 16 + wo];
      a1 += xv * w[(cg + 1) * 16 + wo];
      a2 += xv * w[(cg + 2) * 16 + wo];
      a3 += xv * w[(cg + 3) * 16 + wo];
    }
  }
  const size_t o = (size_t)pix * 64 + cg;
  out[o + 0] = (_Float16)fmaxf(a0, 0.f);
  out[o + 1] = (_Float16)fmaxf(a1, 0.f);
  out[o + 2] = (_Float16)fmaxf(a2, 0.f);
  out[o + 3] = (_Float16)fmaxf(a3, 0.f);
}

// ---------------------------------------------------------------------------
// VQ: per-pixel argmin over 64 codes; q -> f16 NHWC (decoder) + f32 NCHW (qn);
// accumulate sum((q-z)^2) and per-code counts.
// ---------------------------------------------------------------------------
__global__ __launch_bounds__(256) void vq_kernel(
    const float* __restrict__ z, const float* __restrict__ emb,
    _Float16* __restrict__ qf16, float* __restrict__ qn,
    float* __restrict__ sumsq, unsigned* __restrict__ counts) {
  __shared__ float    semb[KCODES * DDIM];
  __shared__ float    esq[KCODES];
  __shared__ unsigned scnt[KCODES];
  __shared__ float    sred[256];

  const int tid = threadIdx.x;
  for (int i = tid; i < KCODES * DDIM; i += 256) semb[i] = emb[i];
  if (tid < KCODES) scnt[tid] = 0u;
  __syncthreads();
  if (tid < KCODES) {
    float s = 0.f;
    for (int d = 0; d < DDIM; ++d) { const float e = semb[tid * DDIM + d]; s += e * e; }
    esq[tid] = s;
  }
  __syncthreads();

  const int m = blockIdx.x * 256 + tid;          // < VQ_ROWS always
  float zr[DDIM];
  const float4* zp = reinterpret_cast<const float4*>(z + (size_t)m * DDIM);
#pragma unroll
  for (int i = 0; i < DDIM / 4; ++i) {
    const float4 t = zp[i];
    zr[4 * i] = t.x; zr[4 * i + 1] = t.y; zr[4 * i + 2] = t.z; zr[4 * i + 3] = t.w;
  }
  float best = 3.4e38f; int bi = 0;
  for (int k = 0; k < KCODES; ++k) {
    const float* e = &semb[k << 6];
    float d0 = 0.f, d1 = 0.f, d2 = 0.f, d3 = 0.f;
#pragma unroll
    for (int i = 0; i < DDIM; i += 4) {
      d0 += zr[i] * e[i];         d1 += zr[i + 1] * e[i + 1];
      d2 += zr[i + 2] * e[i + 2]; d3 += zr[i + 3] * e[i + 3];
    }
    const float s = esq[k] - 2.f * ((d0 + d1) + (d2 + d3));
    if (s < best) { best = s; bi = k; }
  }
  atomicAdd(&scnt[bi], 1u);

  const float* e = &semb[bi << 6];
  float lsum = 0.f;
  const int n  = m >> 12;        // / 4096
  const int yx = m & 4095;
#pragma unroll 8
  for (int d = 0; d < DDIM; ++d) {
    const float q  = e[d];
    const float df = q - zr[d];
    lsum += df * df;
    qf16[(size_t)m * DDIM + d] = (_Float16)q;
    qn[(((size_t)n * DDIM + d) << 12) + yx] = q;   // NCHW straight-through (== q)
  }
  sred[tid] = lsum;
  __syncthreads();
  for (int s = 128; s > 0; s >>= 1) {
    if (tid < s) sred[tid] += sred[tid + s];
    __syncthreads();
  }
  if (tid == 0) atomicAdd(sumsq, sred[0]);
  if (tid < KCODES && scnt[tid]) atomicAdd(&counts[tid], scnt[tid]);
}

__global__ void zero_stats(float* sumsq, unsigned* counts) {
  const int t = threadIdx.x;
  if (t == 0) sumsq[0] = 0.f;
  if (t < KCODES) counts[t] = 0u;
}

__global__ void vq_finalize(const float* sumsq, const unsigned* counts,
                            float* loss_out, float* perp_out) {
  if (threadIdx.x == 0 && blockIdx.x == 0) {
    const float mse = sumsq[0] / ((float)VQ_ROWS * (float)DDIM);
    loss_out[0] = (1.0f + 0.25f) * mse;      // commitment*BETA + embedding
    float ent = 0.f;
    for (int k = 0; k < KCODES; ++k) {
      const float pk = (float)counts[k] / (float)VQ_ROWS;
      ent += pk * logf(pk + 1e-10f);
    }
    perp_out[0] = expf(-ent);
  }
}

// ---------------------------------------------------------------------------
// dec_c3: 64->1ch deconv 4x4 s2 p1 + tanh. Thread per output pixel.
// ---------------------------------------------------------------------------
__global__ void dec_c3_kernel(const _Float16* __restrict__ d2,
                              const float* __restrict__ w,
                              const float* __restrict__ b,
                              float* __restrict__ xre) {
  const int idx = blockIdx.x * blockDim.x + threadIdx.x;   // 2,097,152
  const int n  = idx >> 16;
  const int rm = idx & 65535;
  const int oy = rm >> 8;
  const int ox = rm & 255;
  const int py = oy & 1, px = ox & 1;
  const int yh = oy >> 1, xh = ox >> 1;
  float acc = b[0];
#pragma unroll
  for (int tk = 0; tk < 2; ++tk) {
    const int iy = yh + tk - 1 + py;
    if (iy < 0 || iy >= 128) continue;
    const int ky = py ? (1 + 2 * tk) : (2 * tk);
#pragma unroll
    for (int tx = 0; tx < 2; ++tx) {
      const int ix = xh + tx - 1 + px;
      if (ix < 0 || ix >= 128) continue;
      const int kx = px ? (1 + 2 * tx) : (2 * tx);
      const _Float16* src = d2 + (((size_t)n * 128 + iy) * 128 + ix) * 64;
      const int wo = (3 - ky) * 4 + (3 - kx);
#pragma unroll 8
      for (int c = 0; c < 64; ++c)
        acc += (float)src[c] * w[c * 16 + wo];
    }
  }
  xre[idx] = tanhf(acc);
}

// ---------------------------------------------------------------------------
// Host-side orchestration
// ---------------------------------------------------------------------------
static void launch_gemm(const GemmP& p, hipStream_t s) {
  const int blocks = MTOTAL / 64;
  if (p.Cout == 64)
    gemm_conv_wmma<4><<<blocks, 256, 0, s>>>(p);
  else
    gemm_conv_wmma<8><<<blocks, 256, 0, s>>>(p);
}

extern "C" void kernel_launch(void* const* d_in, const int* in_sizes, int n_in,
                              void* d_out, int out_size, void* d_ws, size_t ws_size,
                              hipStream_t stream) {
  (void)in_sizes; (void)n_in; (void)out_size; (void)ws_size;
  const float* x    = (const float*)d_in[0];
  const float* ec1w = (const float*)d_in[1];
  const float* ec1b = (const float*)d_in[2];
  const float* ec2w = (const float*)d_in[3];
  const float* ec2b = (const float*)d_in[4];
  const float* ec3w = (const float*)d_in[5];
  const float* ec3b = (const float*)d_in[6];
  const float* er1w = (const float*)d_in[7];
  const float* er2w = (const float*)d_in[8];
  const float* vqcw = (const float*)d_in[9];
  const float* vqcb = (const float*)d_in[10];
  const float* emb  = (const float*)d_in[11];
  const float* dc1w = (const float*)d_in[12];
  const float* dc1b = (const float*)d_in[13];
  const float* dr1w = (const float*)d_in[14];
  const float* dr2w = (const float*)d_in[15];
  const float* dc2w = (const float*)d_in[16];
  const float* dc2b = (const float*)d_in[17];
  const float* dc3w = (const float*)d_in[18];
  const float* dc3b = (const float*)d_in[19];
  float* outp = (float*)d_out;

  // ---- workspace layout (bytes) ----
  char* ws = (char*)d_ws;
  _Float16* A1 = (_Float16*)(ws);                 // 67,108,864 B (also dec_c2 out)
  _Float16* A2 = (_Float16*)(ws + 67108864);      // 33,554,432 B (also res hidden H)
  _Float16* X  = (_Float16*)(ws + 100663296);     // 33,554,432 B
  _Float16* X1 = (_Float16*)(ws + 134217728);     // 33,554,432 B
  float*    Z  = (float*)   (ws + 167772160);     // 33,554,432 B
  _Float16* Q  = (_Float16*)(ws + 201326592);     // 16,777,216 B
  _Float16* BP = (_Float16*)(ws + 218103808);     // ~1.31 MB packed weights
  float*    sumsq  = (float*)   (ws + 220200960);
  unsigned* counts = (unsigned*)(ws + 220201216);

  _Float16* bpEncC2 = BP + 0;
  _Float16* bpEncC3 = BP + 131072;
  _Float16* bpEncR1 = BP + 278528;
  _Float16* bpEncR2 = BP + 352256;
  _Float16* bpVqc   = BP + 360448;
  _Float16* bpDecC1 = BP + 368640;
  _Float16* bpDecR1 = BP + 442368;
  _Float16* bpDecR2 = BP + 516096;
  _Float16* bpDecC2 = BP + 524288;                // 4 parity classes x 32768

  // ---- pack all weights (f32 -> fragment-order f16) ----
  auto pack = [&](const float* w, _Float16* o, int Ktot, int Cout, int Cin,
                  int KH, int KW, int mode, int py, int px) {
    const int total = Ktot * Cout;
    pack_b<<<(total + 255) / 256, 256, 0, stream>>>(w, o, Ktot, Cout >> 4,
                                                    Cin, Cout, KH, KW, mode, py, px);
  };
  pack(ec2w, bpEncC2, 1024, 128, 64,  4, 4, 0, 0, 0);
  pack(ec3w, bpEncC3, 1152, 128, 128, 3, 3, 0, 0, 0);
  pack(er1w, bpEncR1, 1152, 64,  128, 3, 3, 0, 0, 0);
  pack(er2w, bpEncR2, 64,   128, 64,  1, 1, 0, 0, 0);
  pack(vqcw, bpVqc,   128,  64,  128, 1, 1, 0, 0, 0);
  pack(dc1w, bpDecC1, 576,  128, 64,  3, 3, 1, 0, 0);
  pack(dr1w, bpDecR1, 1152, 64,  128, 3, 3, 0, 0, 0);
  pack(dr2w, bpDecR2, 64,   128, 64,  1, 1, 0, 0, 0);
  for (int pc = 0; pc < 4; ++pc)
    pack(dc2w, bpDecC2 + pc * 32768, 512, 64, 128, 4, 4, 2, pc >> 1, pc & 1);

  // ---- encoder ----
  enc_c1_kernel<<<32768, 256, 0, stream>>>(x, ec1w, ec1b, A1);

  GemmP p{};
  // enc_c2: (128,128,64) -> (64,64,128), 4x4 s2 p1, relu
  p = GemmP{}; p.A = A1; p.Bp = bpEncC2; p.bias = ec2b; p.Of16 = A2;
  p.IH = 128; p.IW = 128; p.Cin = 64; p.cinsh = 6; p.Cout = 128;
  p.KW = 4; p.stride = 2; p.pad = 1; p.Ktot = 1024; p.relu_out = 1;
  launch_gemm(p, stream);

  // enc_c3: 3x3 s1 p1, no relu -> X
  p = GemmP{}; p.A = A2; p.Bp = bpEncC3; p.bias = ec3b; p.Of16 = X;
  p.IH = 64; p.IW = 64; p.Cin = 128; p.cinsh = 7; p.Cout = 128;
  p.KW = 3; p.stride = 1; p.pad = 1; p.Ktot = 1152;
  launch_gemm(p, stream);

  // encoder residual block (shared weights, 2 iterations)
  for (int it = 0; it < 2; ++it) {
    _Float16* xin  = it ? X1 : X;
    _Float16* xout = it ? X  : X1;
    p = GemmP{}; p.A = xin; p.Bp = bpEncR1; p.Of16 = A2;
    p.IH = 64; p.IW = 64; p.Cin = 128; p.cinsh = 7; p.Cout = 64;
    p.KW = 3; p.stride = 1; p.pad = 1; p.Ktot = 1152; p.relu_a = 1; p.relu_out = 1;
    launch_gemm(p, stream);
    p = GemmP{}; p.A = A2; p.Bp = bpEncR2; p.Res = xin; p.Of16 = xout;
    p.IH = 64; p.IW = 64; p.Cin = 64; p.cinsh = 6; p.Cout = 128;
    p.KW = 1; p.stride = 1; p.pad = 0; p.Ktot = 64;
    launch_gemm(p, stream);
  }

  // pre-VQ 1x1 conv (consumes final relu): X -> z (f32)
  p = GemmP{}; p.A = X; p.Bp = bpVqc; p.bias = vqcb; p.Of32 = Z;
  p.IH = 64; p.IW = 64; p.Cin = 128; p.cinsh = 7; p.Cout = 64;
  p.KW = 1; p.stride = 1; p.pad = 0; p.Ktot = 128; p.relu_a = 1;
  launch_gemm(p, stream);

  // ---- VQ ----
  zero_stats<<<1, 128, 0, stream>>>(sumsq, counts);
  vq_kernel<<<VQ_ROWS / 256, 256, 0, stream>>>(Z, emb, Q, outp + QN_OFF,
                                               sumsq, counts);
  vq_finalize<<<1, 1, 0, stream>>>(sumsq, counts, outp, outp + PERP_OFF);

  // ---- decoder ----
  // dec_c1: deconv 3x3 s1 p1 == conv with flipped kernel (packed mode 1)
  p = GemmP{}; p.A = Q; p.Bp = bpDecC1; p.bias = dc1b; p.Of16 = X;
  p.IH = 64; p.IW = 64; p.Cin = 64; p.cinsh = 6; p.Cout = 128;
  p.KW = 3; p.stride = 1; p.pad = 1; p.Ktot = 576;
  launch_gemm(p, stream);

  for (int it = 0; it < 2; ++it) {
    _Float16* xin  = it ? X1 : X;
    _Float16* xout = it ? X  : X1;
    p = GemmP{}; p.A = xin; p.Bp = bpDecR1; p.Of16 = A2;
    p.IH = 64; p.IW = 64; p.Cin = 128; p.cinsh = 7; p.Cout = 64;
    p.KW = 3; p.stride = 1; p.pad = 1; p.Ktot = 1152; p.relu_a = 1; p.relu_out = 1;
    launch_gemm(p, stream);
    p = GemmP{}; p.A = A2; p.Bp = bpDecR2; p.Res = xin; p.Of16 = xout;
    p.IH = 64; p.IW = 64; p.Cin = 64; p.cinsh = 6; p.Cout = 128;
    p.KW = 1; p.stride = 1; p.pad = 0; p.Ktot = 64;
    launch_gemm(p, stream);
  }

  // dec_c2: deconv 4x4 s2 p1 (consumes final relu), 4 parity classes -> A1 buf
  for (int pc = 0; pc < 4; ++pc) {
    p = GemmP{}; p.A = X; p.Bp = bpDecC2 + pc * 32768; p.bias = dc2b; p.Of16 = A1;
    p.IH = 64; p.IW = 64; p.Cin = 128; p.cinsh = 7; p.Cout = 64;
    p.Ktot = 512; p.mode = 1; p.py = pc >> 1; p.px = pc & 1;
    p.relu_a = 1; p.relu_out = 1;
    launch_gemm(p, stream);
  }

  // dec_c3 + tanh -> x_re
  dec_c3_kernel<<<8192, 256, 0, stream>>>(A1, dc3w, dc3b, outp + XRE_OFF);
}
